// LandmarkRendererGAN_39410619908659
// MI455X (gfx1250) — matrix operand order repaired and checked by reading, not compiled, not run on hardware
//
#include <hip/hip_runtime.h>
#include <hip/hip_bf16.h>
#include <math.h>

typedef __attribute__((ext_vector_type(16))) _Float16 v16h;
typedef __attribute__((ext_vector_type(8)))  float    v8f;
typedef __attribute__((ext_vector_type(4)))  uint32_t u32x4;
typedef __attribute__((ext_vector_type(8)))  uint32_t u32x8;

union V16H { v16h v; _Float16 h[16]; uint32_t u[8]; };

// ---------------- workspace layout (units: f16 halves) ----------------
static constexpr size_t N_X0 = 256ull * 4   * 98 * 98;   // padded input (ident+heatmap)
static constexpr size_t N_A1 = 256ull * 64  * 50 * 50;   // conv1 out, padded
static constexpr size_t N_A2 = 256ull * 128 * 26 * 26;   // conv2 out, padded
static constexpr size_t N_A3 = 256ull * 64  * 50 * 50;   // deconv3 out, padded
static constexpr size_t N_W1 = 4ull  * 2  * 32 * 16;     // packed A frags conv1 (8KB)
static constexpr size_t N_W2 = 8ull  * 32 * 32 * 16;     // conv2 (256KB)
static constexpr size_t N_W3 = 4ull * 4 * 16 * 32 * 16;  // deconv3 (4 classes x 64KB)
static constexpr size_t N_W4 = 4ull * 8 * 32 * 16;       // deconv4 (4 classes x 8KB)

static constexpr size_t OFF_X0 = 0;
static constexpr size_t OFF_A1 = OFF_X0 + N_X0;
static constexpr size_t OFF_A2 = OFF_A1 + N_A1;
static constexpr size_t OFF_A3 = OFF_A2 + N_A2;
static constexpr size_t OFF_W1 = OFF_A3 + N_A3;
static constexpr size_t OFF_W2 = OFF_W1 + N_W1;
static constexpr size_t OFF_W3 = OFF_W2 + N_W2;
static constexpr size_t OFF_W4 = OFF_W3 + N_W3;
static constexpr size_t TOTAL_HALVES = OFF_W4 + N_W4;    // ~228 MB

// k order used here for A and B (contiguous): lane<16 -> elems 0..15 = K0..15
//                                  lane>=16 -> elems 0..15 = K16..31
__host__ __device__ inline int k32map(int lane, int e) {
    return e | ((lane >> 4) << 4);
}

__device__ inline v8f wmma_f16(v16h a, v16h b, v8f c) {
    return __builtin_amdgcn_wmma_f32_16x16x32_f16(false, a, false, b, (short)0, c,
                                                  false, false);
}

// Stage `nbytes` from global into LDS with per-lane async copies (ASYNCcnt path).
__device__ inline void stage_to_lds(_Float16* sdst, const _Float16* gsrc, int nbytes) {
    const uint32_t lbase = (uint32_t)(uintptr_t)sdst;   // low 32b of LDS aperture = offset
    const char* gs = (const char*)gsrc;
    for (int it = threadIdx.x; it < (nbytes >> 4); it += blockDim.x) {
        uint32_t loff = lbase + (it << 4);
        const void* ga = gs + (it << 4);
        asm volatile("global_load_async_to_lds_b128 %0, %1, off"
                     :: "v"(loff), "v"(ga) : "memory");
    }
    asm volatile("s_wait_asynccnt 0x0" ::: "memory");
    __syncthreads();
}

// Stage a contiguous `nbytes` block (multiple of 8) via the Tensor Data Mover:
// one DMA descriptor (1 row of nbytes/8 8-byte elements), issued by wave 0 only,
// tracked with TENSORcnt (CDNA5 ISA-8).
__device__ inline void tdm_stage_to_lds(_Float16* sdst, const _Float16* gsrc,
                                        uint32_t nbytes) {
    if (threadIdx.x < 32) {                             // wave-uniform branch
        const uint64_t ga  = (uint64_t)(uintptr_t)gsrc;
        const uint32_t lds = (uint32_t)(uintptr_t)sdst;
        const uint32_t n8  = nbytes >> 3;               // 8-byte elements per row
        u32x4 g0;
        g0[0] = 1u;                                     // count=1 valid user D#
        g0[1] = lds;                                    // lds_addr
        g0[2] = (uint32_t)ga;                           // global_addr[31:0]
        g0[3] = (uint32_t)(ga >> 32) | (2u << 30);      // global_addr[56:32] | type=2
        u32x8 g1;
        g1[0] = 3u << 16;                               // data_size=8B; mask/flags=0
        g1[1] = (n8 & 0xFFFFu) << 16;                   // tensor_dim0[15:0]
        g1[2] = (n8 >> 16) | (1u << 16);                // tensor_dim0[31:16] | tensor_dim1=1
        g1[3] = (n8 & 0xFFFFu) << 16;                   // tile_dim0
        g1[4] = 1u;                                     // tile_dim1=1, tile_dim2=0
        g1[5] = n8;                                     // tensor_dim0_stride[31:0]
        g1[6] = 0u;
        g1[7] = 0u;
        asm volatile("tensor_load_to_lds %0, %1" :: "s"(g0), "s"(g1) : "memory");
        __builtin_amdgcn_s_wait_tensorcnt(0);
    }
    __syncthreads();
}

// ---------------- utility kernels ----------------
__global__ __launch_bounds__(256) void k_zero(uint4* p, size_t n) {
    size_t i = (size_t)blockIdx.x * 256 + threadIdx.x;
    if (i < n) p[i] = make_uint4(0u, 0u, 0u, 0u);
}

// pack w1 [64,4,4,4] -> A frags: ((mt*2+ks)*32+lane)*16+e
__global__ __launch_bounds__(256) void k_pack1(_Float16* ws, const float* w1) {
    int idx = blockIdx.x * 256 + threadIdx.x;           // < 4096
    int e = idx & 15, lane = (idx >> 4) & 31, ks = (idx >> 9) & 1, mt = idx >> 10;
    int m = mt * 16 + (lane & 15);
    int K = ks * 32 + k32map(lane, e);                  // K = c*16 + ky*4 + kx
    int c = K >> 4, t = K & 15, ky = t >> 2, kx = t & 3;
    ws[OFF_W1 + idx] = (_Float16)w1[((m * 4 + c) * 4 + ky) * 4 + kx];
}

// pack w2 [128,64,4,4]
__global__ __launch_bounds__(256) void k_pack2(_Float16* ws, const float* w2) {
    int idx = blockIdx.x * 256 + threadIdx.x;           // < 131072
    int e = idx & 15, lane = (idx >> 4) & 31, ks = (idx >> 9) & 31, mt = idx >> 14;
    int m = mt * 16 + (lane & 15);
    int K = ks * 32 + k32map(lane, e);
    int c = K >> 4, t = K & 15, ky = t >> 2, kx = t & 3;
    ws[OFF_W2 + idx] = (_Float16)w2[((m * 64 + c) * 4 + ky) * 4 + kx];
}

// pack w3 [128,64,4,4] (ConvTranspose [I,O,kh,kw]) into 4 parity-class GEMMs
__global__ __launch_bounds__(256) void k_pack3(_Float16* ws, const float* w3) {
    int idx = blockIdx.x * 256 + threadIdx.x;           // < 131072
    int e = idx & 15, lane = (idx >> 4) & 31, ks = (idx >> 9) & 15;
    int mt = (idx >> 13) & 3, cls = idx >> 15;
    int ry = cls >> 1, rx = cls & 1;
    int K = ks * 32 + k32map(lane, e);                  // K = c*4 + tky*2 + tkx
    int c = K >> 2, t = K & 3, tky = t >> 1, tkx = t & 1;
    int ky = 2 * tky + (1 - ry), kx = 2 * tkx + (1 - rx);
    int o = mt * 16 + (lane & 15);
    ws[OFF_W3 + idx] = (_Float16)w3[((c * 64 + o) * 4 + ky) * 4 + kx];
}

// pack w4 [64,3,4,4]; pad out-channel rows 3..15 with zero
__global__ __launch_bounds__(256) void k_pack4(_Float16* ws, const float* w4) {
    int idx = blockIdx.x * 256 + threadIdx.x;           // < 16384
    int e = idx & 15, lane = (idx >> 4) & 31, ks = (idx >> 9) & 7, cls = idx >> 12;
    int ry = cls >> 1, rx = cls & 1;
    int K = ks * 32 + k32map(lane, e);                  // K = c*4 + tky*2 + tkx
    int c = K >> 2, t = K & 3, tky = t >> 1, tkx = t & 1;
    int ky = 2 * tky + (1 - ry), kx = 2 * tkx + (1 - rx);
    int o = lane & 15;
    float v = (o < 3) ? w4[((c * 3 + o) * 4 + ky) * 4 + kx] : 0.0f;
    ws[OFF_W4 + idx] = (_Float16)v;
}

// broadcast identity image into channels 0..2 of x0 (padded 98x98, +1 halo)
__global__ __launch_bounds__(256) void k_ident(_Float16* ws, const float* ident) {
    int idx = blockIdx.x * 256 + threadIdx.x;           // < 256*3*96*96
    int x = idx % 96; int t = idx / 96;
    int y = t % 96;   t /= 96;
    int c = t % 3;    int f = t / 3;
    int b = f >> 5;                                     // T = 32
    float v = ident[(((size_t)b * 3 + c) * 96 + y) * 96 + x];
    ws[OFF_X0 + ((size_t)(f * 4 + c)) * 9604 + (y + 1) * 98 + (x + 1)] = (_Float16)v;
}

// landmark scatter into channel 3
__global__ __launch_bounds__(256) void k_scatter(_Float16* ws, const float* lm) {
    int idx = blockIdx.x * 256 + threadIdx.x;           // 256*68 exactly
    int f = idx / 68, p = idx % 68;
    float fx = lm[(size_t)f * 136 + p * 2 + 0] * 95.0f;
    float fy = lm[(size_t)f * 136 + p * 2 + 1] * 95.0f;
    int x = (int)fx, y = (int)fy;                       // trunc toward zero, like .long()
    if (x >= 0 && x < 96 && y >= 0 && y < 96)
        ws[OFF_X0 + ((size_t)(f * 4 + 3)) * 9604 + (y + 1) * 98 + (x + 1)] = (_Float16)1.0f;
}

// ---------------- conv1: 4->64, stride 2, K=64 (2 wmma ksteps), N=256*48*48 ----------------
__global__ __launch_bounds__(256) void k_conv1(_Float16* ws, const float* bias) {
    extern __shared__ _Float16 sW1[];                   // 8 KB (ASYNCcnt path)
    stage_to_lds(sW1, ws + OFF_W1, (int)(N_W1 * 2));
    const int wid = threadIdx.x >> 5, lane = threadIdx.x & 31;
    const int col = lane & 15, khi = lane >> 4;
    const int ntile = blockIdx.x * 8 + wid;             // 36864 tiles
    const int n = ntile * 16 + col;
    const int f = n / 2304, rem = n % 2304;
    const int oy = rem / 48, ox = rem % 48;
    const _Float16* x0 = ws + OFF_X0;
    _Float16* a1 = ws + OFF_A1;
    v8f acc[4] = {};
#pragma unroll
    for (int ks = 0; ks < 2; ++ks) {
        int c = (ks * 32 + khi * 16) >> 4;              // one 4x4 patch per 16-K group
        const _Float16* pc = x0 + ((size_t)(f * 4 + c)) * 9604;
        V16H b;
#pragma unroll
        for (int ky = 0; ky < 4; ++ky) {
            const uint32_t* pr = (const uint32_t*)(pc + (2 * oy + ky) * 98 + 2 * ox);
            b.u[2 * ky + 0] = pr[0];
            b.u[2 * ky + 1] = pr[1];
        }
#pragma unroll
        for (int mt = 0; mt < 4; ++mt) {
            v16h a = *(const v16h*)(sW1 + ((((mt * 2 + ks) * 32) + lane) << 4));
            acc[mt] = wmma_f16(a, b.v, acc[mt]);
        }
    }
#pragma unroll
    for (int mt = 0; mt < 4; ++mt)
#pragma unroll
        for (int r = 0; r < 8; ++r) {
            int m = mt * 16 + r + khi * 8;
            float v = acc[mt][r] + bias[m];
            v = v > 0.0f ? v : 0.0f;
            a1[((size_t)(f * 64 + m)) * 2500 + (oy + 1) * 50 + (ox + 1)] = (_Float16)v;
        }
}

// -------- conv2: 64->128, stride 2, K=1024 (32 ksteps), N=256*24*24 --------
// blockIdx.y = M-group (2 mtiles); 64 KB weight slice DMA'd to LDS by the TDM.
__global__ __launch_bounds__(256) void k_conv2(_Float16* ws, const float* bias) {
    extern __shared__ _Float16 sW2[];                   // 64 KB
    const int mg = blockIdx.y;                          // 0..3
    tdm_stage_to_lds(sW2, ws + OFF_W2 + (size_t)mg * (2 * 32 * 32 * 16),
                     2 * 32 * 32 * 16 * 2);
    const int wid = threadIdx.x >> 5, lane = threadIdx.x & 31;
    const int col = lane & 15, khi = lane >> 4;
    const int ntile = blockIdx.x * 8 + wid;             // 9216 tiles
    const int n = ntile * 16 + col;
    const int f = n / 576, rem = n % 576;
    const int oy = rem / 24, ox = rem % 24;
    const _Float16* a1 = ws + OFF_A1;
    _Float16* a2 = ws + OFF_A2;
    v8f acc[2] = {};
    for (int ks = 0; ks < 32; ++ks) {
        int c = (ks * 32 + khi * 16) >> 4;
        const _Float16* pc = a1 + ((size_t)(f * 64 + c)) * 2500;
        __builtin_prefetch((const void*)(pc + 2 * 2500), 0, 3);   // next activation plane
        V16H b;
#pragma unroll
        for (int ky = 0; ky < 4; ++ky) {
            const uint32_t* pr = (const uint32_t*)(pc + (2 * oy + ky) * 50 + 2 * ox);
            b.u[2 * ky + 0] = pr[0];
            b.u[2 * ky + 1] = pr[1];
        }
#pragma unroll
        for (int mtl = 0; mtl < 2; ++mtl) {
            v16h a = *(const v16h*)(sW2 + ((((mtl * 32 + ks) * 32) + lane) << 4));
            acc[mtl] = wmma_f16(a, b.v, acc[mtl]);
        }
    }
#pragma unroll
    for (int mtl = 0; mtl < 2; ++mtl)
#pragma unroll
        for (int r = 0; r < 8; ++r) {
            int m = (mg * 2 + mtl) * 16 + r + khi * 8;
            float v = acc[mtl][r] + bias[m];
            v = v > 0.0f ? v : 0.0f;
            a2[((size_t)(f * 128 + m)) * 676 + (oy + 1) * 26 + (ox + 1)] = (_Float16)v;
        }
}

// ---------------- deconv3: 128->64, parity-class GEMM, K=512 (16 ksteps) ----------------
__global__ __launch_bounds__(256) void k_deconv3(_Float16* ws, const float* bias) {
    extern __shared__ _Float16 sW3[];                   // 64 KB (one parity class, TDM)
    const int cls = blockIdx.y, ry = cls >> 1, rx = cls & 1;
    tdm_stage_to_lds(sW3, ws + OFF_W3 + (size_t)cls * (4 * 16 * 32 * 16),
                     4 * 16 * 32 * 16 * 2);
    const int wid = threadIdx.x >> 5, lane = threadIdx.x & 31;
    const int col = lane & 15, khi = lane >> 4;
    const int ntile = blockIdx.x * 8 + wid;             // 9216 tiles/class
    const int n = ntile * 16 + col;
    const int f = n / 576, rem = n % 576;
    const int j = rem / 24, i = rem % 24;
    const _Float16* a2 = ws + OFF_A2;
    _Float16* a3 = ws + OFF_A3;
    v8f acc[4] = {};
    for (int ks = 0; ks < 16; ++ks) {
        int c0 = (ks * 32 + khi * 16) >> 2;             // 4 channels x 4 taps per 16-K group
        V16H b;
#pragma unroll
        for (int g = 0; g < 4; ++g) {
            const _Float16* pc = a2 + ((size_t)(f * 128 + c0 + g)) * 676;
#pragma unroll
            for (int t = 0; t < 4; ++t) {
                int tky = t >> 1, tkx = t & 1;
                int row = j + ry - tky + 1;             // halo handles +-1
                int cx  = i + rx - tkx + 1;
                b.h[g * 4 + t] = pc[row * 26 + cx];
            }
        }
#pragma unroll
        for (int mt = 0; mt < 4; ++mt) {
            v16h a = *(const v16h*)(sW3 + ((((mt * 16 + ks) * 32) + lane) << 4));
            acc[mt] = wmma_f16(a, b.v, acc[mt]);
        }
    }
    const int y = 2 * j + ry, x = 2 * i + rx;
#pragma unroll
    for (int mt = 0; mt < 4; ++mt)
#pragma unroll
        for (int r = 0; r < 8; ++r) {
            int m = mt * 16 + r + khi * 8;
            float v = acc[mt][r] + bias[m];
            v = v > 0.0f ? v : 0.0f;
            a3[((size_t)(f * 64 + m)) * 2500 + (y + 1) * 50 + (x + 1)] = (_Float16)v;
        }
}

// ---------------- deconv4: 64->3 + tanh, parity-class GEMM, K=256 (8 ksteps) ----------------
__global__ __launch_bounds__(256) void k_deconv4(_Float16* ws, const float* bias,
                                                 float* out) {
    extern __shared__ _Float16 sW4[];                   // 8 KB (ASYNCcnt path)
    const int cls = blockIdx.y, ry = cls >> 1, rx = cls & 1;
    stage_to_lds(sW4, ws + OFF_W4 + (size_t)cls * (8 * 32 * 16), 8 * 32 * 16 * 2);
    const int wid = threadIdx.x >> 5, lane = threadIdx.x & 31;
    const int col = lane & 15, khi = lane >> 4;
    const int ntile = blockIdx.x * 8 + wid;             // 36864 tiles/class
    const int n = ntile * 16 + col;
    const int f = n / 2304, rem = n % 2304;
    const int j = rem / 48, i = rem % 48;
    const _Float16* a3 = ws + OFF_A3;
    v8f acc = {};
    for (int ks = 0; ks < 8; ++ks) {
        int c0 = (ks * 32 + khi * 16) >> 2;
        V16H b;
#pragma unroll
        for (int g = 0; g < 4; ++g) {
            const _Float16* pc = a3 + ((size_t)(f * 64 + c0 + g)) * 2500;
#pragma unroll
            for (int t = 0; t < 4; ++t) {
                int tky = t >> 1, tkx = t & 1;
                int row = j + ry - tky + 1;
                int cx  = i + rx - tkx + 1;
                b.h[g * 4 + t] = pc[row * 50 + cx];
            }
        }
        v16h a = *(const v16h*)(sW4 + (((ks * 32) + lane) << 4));
        acc = wmma_f16(a, b.v, acc);
    }
    const int y = 2 * j + ry, x = 2 * i + rx;
    if (khi == 0) {                                     // lanes 0..15 hold M=0..7
#pragma unroll
        for (int r = 0; r < 3; ++r) {
            float v = tanhf(acc[r] + bias[r]);
            out[((size_t)f * 3 + r) * 9216 + y * 96 + x] = v;
        }
    }
}

extern "C" void kernel_launch(void* const* d_in, const int* in_sizes, int n_in,
                              void* d_out, int out_size, void* d_ws, size_t ws_size,
                              hipStream_t stream) {
    const float* identity = (const float*)d_in[0];
    const float* lmk      = (const float*)d_in[1];
    const float* w1 = (const float*)d_in[2];
    const float* b1 = (const float*)d_in[3];
    const float* w2 = (const float*)d_in[4];
    const float* b2 = (const float*)d_in[5];
    const float* w3 = (const float*)d_in[6];
    const float* b3 = (const float*)d_in[7];
    const float* w4 = (const float*)d_in[8];
    const float* b4 = (const float*)d_in[9];
    float* out = (float*)d_out;
    _Float16* ws = (_Float16*)d_ws;

    const size_t nz = TOTAL_HALVES / 8;                 // uint4 chunks
    k_zero<<<(unsigned)((nz + 255) / 256), 256, 0, stream>>>((uint4*)d_ws, nz);
    k_pack1<<<16,  256, 0, stream>>>(ws, w1);
    k_pack2<<<512, 256, 0, stream>>>(ws, w2);
    k_pack3<<<512, 256, 0, stream>>>(ws, w3);
    k_pack4<<<64,  256, 0, stream>>>(ws, w4);
    k_ident<<<27648, 256, 0, stream>>>(ws, identity);
    k_scatter<<<68,  256, 0, stream>>>(ws, lmk);
    k_conv1<<<4608, 256, 8192, stream>>>(ws, b1);
    k_conv2<<<dim3(1152, 4), 256, 65536, stream>>>(ws, b2);
    k_deconv3<<<dim3(1152, 4), 256, 65536, stream>>>(ws, b3);
    k_deconv4<<<dim3(4608, 4), 256, 8192, stream>>>(ws, b4, out);
}
